// CosSimNegHead_64218351010165
// MI455X (gfx1250) — matrix operand order repaired and checked by reading, not compiled, not run on hardware
//
#include <hip/hip_runtime.h>
#include <hip/hip_bf16.h>

// ---------------------------------------------------------------------------
// CosSimNegHead on MI455X (gfx1250): f16 WMMA GEMMs with fp32 accumulate,
// TDM (tensor_load_to_lds) double-buffered tile staging.
// ---------------------------------------------------------------------------

typedef __attribute__((ext_vector_type(16))) _Float16 v16h;
typedef __attribute__((ext_vector_type(8)))  _Float16 v8h;
typedef __attribute__((ext_vector_type(4)))  _Float16 v4h;
typedef __attribute__((ext_vector_type(8)))  float    v8f;
typedef __attribute__((ext_vector_type(4)))  unsigned int u32x4;
typedef __attribute__((ext_vector_type(8)))  int      i32x8;
typedef __attribute__((ext_vector_type(4)))  int      i32x4;

#define TILE_M 128
#define TILE_N 128
#define TILE_K 32
#define LDT    40   // LDS row stride in halves (80B): 64B data row + 16B TDM pad

union V16U { v16h v; v8h h[2]; };

// ---------------------------------------------------------------------------
// TDM 2D tile load: global (row-major, elem=2B) -> LDS with row padding.
// D# packing per CDNA5 ISA.
//   pad_interval=3 -> every 16 DWORDs (= tile_dim0=32 halves = 64B row)
//   pad_amount  =3 -> 4 DWORDs (8 halves) => LDS row stride = 40 halves = LDT
// This toolchain's builtin takes 6 args: (g0, g1, g2, g3, g_extra, cpol).
// ---------------------------------------------------------------------------
__device__ __forceinline__ void tdm_load_tile_f16(
    unsigned lds_addr, const _Float16* gaddr,
    unsigned tile_w, unsigned tile_rows,
    unsigned tensor_w, unsigned tensor_rows, unsigned row_stride_elems) {
    unsigned long long ga = (unsigned long long)(uintptr_t)gaddr;
    u32x4 g0;
    g0[0] = 1u;                                    // count=1 (valid), user desc
    g0[1] = lds_addr;                              // LDS byte address
    g0[2] = (unsigned)(ga & 0xFFFFFFFFu);          // global_addr[31:0]
    g0[3] = (unsigned)((ga >> 32) & 0x01FFFFFFu)   // global_addr[56:32]
          | (2u << 30);                            // type=2 ("image")
    i32x8 g1;
    g1[0] = (int)((1u << 16)    // data_size=1 (2 bytes)
          |       (1u << 20)    // pad_enable
          |       (3u << 22)    // pad_interval: 16 DWORDs
          |       (3u << 25));  // pad_amount: 4 DWORDs
    g1[1] = (int)((tensor_w & 0xFFFFu) << 16);                         // dim0 lo16
    g1[2] = (int)(((tensor_w >> 16) & 0xFFFFu)
          |       ((tensor_rows & 0xFFFFu) << 16));                    // dim0 hi / dim1 lo
    g1[3] = (int)(((tensor_rows >> 16) & 0xFFFFu)
          |       ((tile_w & 0xFFFFu) << 16));                         // dim1 hi / tile_dim0
    g1[4] = (int)(tile_rows & 0xFFFFu);                                // tile_dim1, tile_dim2=0
    g1[5] = (int)row_stride_elems;                                     // dim0_stride lo32
    g1[6] = 0;                                                         // dim0_stride hi / dim1_stride lo
    g1[7] = 0;
    i32x4 z4 = {0, 0, 0, 0};
    i32x8 z8 = {0, 0, 0, 0, 0, 0, 0, 0};
    __builtin_amdgcn_tensor_load_to_lds(g0, g1, z4, z4, z8, 0);
}

// ---------------------------------------------------------------------------
// f32 -> f16 conversion, 4 elements/thread/iter
// ---------------------------------------------------------------------------
__global__ __launch_bounds__(256)
void cvt_f32_f16_kernel(const float* __restrict__ in, _Float16* __restrict__ out, long n4) {
    long i = (long)blockIdx.x * blockDim.x + threadIdx.x;
    long stride = (long)gridDim.x * blockDim.x;
    for (; i < n4; i += stride) {
        float4 v = ((const float4*)in)[i];
        v4h o = { (_Float16)v.x, (_Float16)v.y, (_Float16)v.z, (_Float16)v.w };
        ((v4h*)out)[i] = o;
    }
}

// ---------------------------------------------------------------------------
// per-row 1/max(||x||,eps); one block per row
// ---------------------------------------------------------------------------
__global__ __launch_bounds__(256)
void rownorm_kernel(const float* __restrict__ x, float* __restrict__ invn, int K) {
    __shared__ float red[256];
    const int row = blockIdx.x;
    const float* p = x + (size_t)row * K;
    float s = 0.0f;
    for (int i = threadIdx.x; i < K; i += 256) { float v = p[i]; s += v * v; }
    red[threadIdx.x] = s;
    __syncthreads();
    for (int off = 128; off > 0; off >>= 1) {
        if (threadIdx.x < off) red[threadIdx.x] += red[threadIdx.x + off];
        __syncthreads();
    }
    if (threadIdx.x == 0) invn[row] = 1.0f / fmaxf(sqrtf(red[0]), 1e-8f);
}

// ---------------------------------------------------------------------------
// normalize 60 reps + 180 neg reps into one f16 matrix R (240 x K)
// ---------------------------------------------------------------------------
__global__ __launch_bounds__(256)
void repnorm_kernel(const float* __restrict__ reps, const float* __restrict__ negreps,
                    _Float16* __restrict__ R, int K) {
    __shared__ float red[256];
    __shared__ float inv;
    const int row = blockIdx.x;  // 0..239
    const float* p = (row < 60) ? (reps + (size_t)row * K)
                                : (negreps + (size_t)(row - 60) * K);
    float s = 0.0f;
    for (int i = threadIdx.x; i < K; i += 256) { float v = p[i]; s += v * v; }
    red[threadIdx.x] = s;
    __syncthreads();
    for (int off = 128; off > 0; off >>= 1) {
        if (threadIdx.x < off) red[threadIdx.x] += red[threadIdx.x + off];
        __syncthreads();
    }
    if (threadIdx.x == 0) inv = 1.0f / fmaxf(sqrtf(red[0]), 1e-8f);
    __syncthreads();
    const float iv = inv;
    for (int i = threadIdx.x; i < K; i += 256)
        R[(size_t)row * K + i] = (_Float16)(p[i] * iv);
}

// ---------------------------------------------------------------------------
// Tiled f16 WMMA GEMM: out[M,N] = A[M,K] @ B[N,K]^T (+bias[col]) (*rowscale[row])
// Block: 256 threads = 8 waves as 4(M) x 2(N); wave tile 32x64 = 2x4 WMMA tiles.
// TDM double-buffered LDS staging; wave 0 drives the DMA.
// ---------------------------------------------------------------------------
__global__ __launch_bounds__(256)
void gemm_f16_wmma_kernel(const _Float16* __restrict__ A, int lda,
                          const _Float16* __restrict__ B, int ldb,
                          int M, int Nn, int K,
                          const float* __restrict__ bias,
                          const float* __restrict__ rowscale,
                          _Float16* __restrict__ outH,
                          float* __restrict__ outF, int ldo) {
    __shared__ alignas(16) _Float16 Alds[2][TILE_M * LDT];
    __shared__ alignas(16) _Float16 Blds[2][TILE_N * LDT];

    const int tid   = threadIdx.x;
    const int lane  = tid & 31;
    const int wid   = tid >> 5;      // 0..7
    const int waveM = wid & 3;       // 4 waves along M -> 4*32 = 128
    const int waveN = wid >> 2;      // 2 waves along N -> 2*64 = 128
    const int m0    = blockIdx.y * TILE_M;
    const int n0    = blockIdx.x * TILE_N;

    const int mlane = lane & 15;
    const int hi    = lane >> 4;     // 0: lanes 0-15, 1: lanes 16-31

    v8f acc[2][4] = {};

    const int nIter = K / TILE_K;

    // prologue: stage tile 0 into buffer 0
    if (wid == 0) {
        tdm_load_tile_f16((unsigned)(uintptr_t)(void*)&Alds[0][0],
                          A + (size_t)m0 * lda, TILE_K, TILE_M,
                          (unsigned)K, (unsigned)(M - m0), (unsigned)lda);
        tdm_load_tile_f16((unsigned)(uintptr_t)(void*)&Blds[0][0],
                          B + (size_t)n0 * ldb, TILE_K, TILE_N,
                          (unsigned)K, (unsigned)(Nn - n0), (unsigned)ldb);
    }

    for (int it = 0; it < nIter; ++it) {
        const int buf = it & 1;
        if (wid == 0) {
            if (it + 1 < nIter) {
                const int k1 = (it + 1) * TILE_K;
                tdm_load_tile_f16((unsigned)(uintptr_t)(void*)&Alds[buf ^ 1][0],
                                  A + (size_t)m0 * lda + k1, TILE_K, TILE_M,
                                  (unsigned)(K - k1), (unsigned)(M - m0), (unsigned)lda);
                tdm_load_tile_f16((unsigned)(uintptr_t)(void*)&Blds[buf ^ 1][0],
                                  B + (size_t)n0 * ldb + k1, TILE_K, TILE_N,
                                  (unsigned)(K - k1), (unsigned)(Nn - n0), (unsigned)ldb);
                __builtin_amdgcn_s_wait_tensorcnt(2);  // current tile done, next in flight
            } else {
                __builtin_amdgcn_s_wait_tensorcnt(0);
            }
        }
        __syncthreads();   // release consumers once wave0's TDM for `buf` completed

        // ---- fragments per CDNA5 ISA 16-bit layouts ----
        // A 16x32: lane m (hi=0) holds K {0-7,16-23}; lane m+16 (hi=1) K {8-15,24-31}
        v16h afrag[2];
#pragma unroll
        for (int mi = 0; mi < 2; ++mi) {
            const _Float16* p = &Alds[buf][(waveM * 32 + mi * 16 + mlane) * LDT + hi * 8];
            V16U u;
            u.h[0] = *(const v8h*)(p);
            u.h[1] = *(const v8h*)(p + 16);
            afrag[mi] = u.v;
        }
        // B 32x16: lane n (hi=0) holds K {0-7,16-23} of column n; lane n+16 (hi=1) K {8-15,24-31}
        v16h bfrag[4];
#pragma unroll
        for (int ni = 0; ni < 4; ++ni) {
            const _Float16* p = &Blds[buf][(waveN * 64 + ni * 16 + mlane) * LDT + hi * 8];
            V16U u;
            u.h[0] = *(const v8h*)(p);
            u.h[1] = *(const v8h*)(p + 16);
            bfrag[ni] = u.v;
        }

#pragma unroll
        for (int mi = 0; mi < 2; ++mi)
#pragma unroll
            for (int ni = 0; ni < 4; ++ni)
                acc[mi][ni] = __builtin_amdgcn_wmma_f32_16x16x32_f16(
                    false, afrag[mi], false, bfrag[ni],
                    (short)0, acc[mi][ni], false, false);

        __syncthreads();   // everyone done reading `buf` before TDM overwrites it
    }

    // ---- epilogue: D layout => VGPR j, lane l: row = j + 8*hi, col = l&15 ----
#pragma unroll
    for (int mi = 0; mi < 2; ++mi) {
#pragma unroll
        for (int ni = 0; ni < 4; ++ni) {
            const int col = n0 + waveN * 64 + ni * 16 + mlane;
            if (col >= Nn) continue;
            const float bv = bias ? bias[col] : 0.0f;
#pragma unroll
            for (int j = 0; j < 8; ++j) {
                const int row = m0 + waveM * 32 + mi * 16 + hi * 8 + j;
                float v = acc[mi][ni][j] + bv;
                if (rowscale) v *= rowscale[row];
                if (outH) outH[(size_t)row * ldo + col] = (_Float16)v;
                else      outF[(size_t)row * ldo + col] = v;
            }
        }
    }
}

// ---------------------------------------------------------------------------
// Epilogue: scores / cos_sim / neg max / neg_scores / bg pad
// ---------------------------------------------------------------------------
__global__ __launch_bounds__(64)
void cos_epilogue_kernel(const float* __restrict__ cos_all,
                         float* __restrict__ scores,
                         float* __restrict__ cos_sim,
                         float* __restrict__ neg_cos,
                         float* __restrict__ neg_scores) {
    const int row = blockIdx.x;
    const int c = threadIdx.x;
    const float* pr = cos_all + (size_t)row * 240;
    if (c < 60) {
        float cs = pr[c];
        cos_sim[(size_t)row * 60 + c] = cs;
        scores[(size_t)row * 61 + c]  = expf(-(1.0f - cs) * 20.0f);
        float m = pr[60 + c * 3];
        m = fmaxf(m, pr[60 + c * 3 + 1]);
        m = fmaxf(m, pr[60 + c * 3 + 2]);
        neg_cos[(size_t)row * 60 + c]    = m;
        neg_scores[(size_t)row * 60 + c] = expf(-(1.0f - m) * 20.0f);
    } else if (c == 60) {
        scores[(size_t)row * 61 + 60] = -100000.0f;
    }
}

// ---------------------------------------------------------------------------
extern "C" void kernel_launch(void* const* d_in, const int* in_sizes, int n_in,
                              void* d_out, int out_size, void* d_ws, size_t ws_size,
                              hipStream_t stream) {
    (void)in_sizes; (void)n_in; (void)out_size; (void)ws_size;

    const float* x         = (const float*)d_in[0];  // (16384, 2048)
    const float* reps_w    = (const float*)d_in[1];  // (61, 2048)
    const float* neg_reps  = (const float*)d_in[2];  // (180, 2048)
    const float* reg_fc_w  = (const float*)d_in[3];  // (2048, 2048)
    const float* reg_fc_b  = (const float*)d_in[4];  // (2048,)
    const float* bbox_w    = (const float*)d_in[5];  // (240, 2048)
    const float* bbox_b    = (const float*)d_in[6];  // (240,)

    constexpr int N  = 16384;
    constexpr int D  = 2048;
    constexpr int H  = 2048;
    constexpr int NB = 240;   // 60 classes * 4 box / also 60 + 180 rep rows

    float* out        = (float*)d_out;
    float* scores     = out;                               // N*61
    float* deltas     = scores + (size_t)N * 61;           // N*240
    float* cos_sim    = deltas + (size_t)N * 240;          // N*60
    float* neg_cos    = cos_sim + (size_t)N * 60;          // N*60
    float* neg_scores = neg_cos + (size_t)N * 60;          // N*60

    // workspace carve-up (256B aligned)
    char*  ws  = (char*)d_ws;
    size_t off = 0;
    auto carve = [&](size_t bytes) -> char* {
        char* p = ws + off;
        off = (off + bytes + 255) & ~(size_t)255;
        return p;
    };
    _Float16* x16    = (_Float16*)carve((size_t)N * D * 2);
    _Float16* h16    = (_Float16*)carve((size_t)N * H * 2);
    _Float16* w16    = (_Float16*)carve((size_t)H * D * 2);
    _Float16* bbox16 = (_Float16*)carve((size_t)NB * H * 2);
    _Float16* R16    = (_Float16*)carve((size_t)NB * D * 2);
    float*    invn   = (float*)carve((size_t)N * 4);
    float*    cosall = (float*)carve((size_t)N * NB * 4);

    // 1) precision downconverts
    cvt_f32_f16_kernel<<<4096, 256, 0, stream>>>(x, x16, (long)N * D / 4);
    cvt_f32_f16_kernel<<<2048, 256, 0, stream>>>(reg_fc_w, w16, (long)H * D / 4);
    cvt_f32_f16_kernel<<<512, 256, 0, stream>>>(bbox_w, bbox16, (long)NB * H / 4);

    // 2) row inverse norms of x; 3) normalized rep matrix (60 reps + 180 negs)
    rownorm_kernel<<<N, 256, 0, stream>>>(x, invn, D);
    repnorm_kernel<<<NB, 256, 0, stream>>>(reps_w, neg_reps, R16, D);

    // 4) h = x @ reg_fc_w^T + b  (f16 out, 137 GFLOP)
    gemm_f16_wmma_kernel<<<dim3(H / TILE_N, N / TILE_M), 256, 0, stream>>>(
        x16, D, w16, D, N, H, D, reg_fc_b, nullptr, h16, nullptr, H);

    // 5) deltas = h @ bbox_w^T + b  (f32 out)
    gemm_f16_wmma_kernel<<<dim3((NB + TILE_N - 1) / TILE_N, N / TILE_M), 256, 0, stream>>>(
        h16, H, bbox16, H, N, NB, H, bbox_b, nullptr, nullptr, deltas, NB);

    // 6) cos_all = (x @ R^T) * invnorm(row)  (f32 out)
    gemm_f16_wmma_kernel<<<dim3((NB + TILE_N - 1) / TILE_N, N / TILE_M), 256, 0, stream>>>(
        x16, D, R16, D, N, NB, D, nullptr, invn, nullptr, cosall, NB);

    // 7) scores / neg max / exps / bg pad
    cos_epilogue_kernel<<<N, 64, 0, stream>>>(cosall, scores, cos_sim, neg_cos, neg_scores);
}
